// MyRNN_49684181680838
// MI455X (gfx1250) — matrix operand order repaired and checked by reading, not compiled, not run on hardware
//
#include <hip/hip_runtime.h>
#include <hip/hip_bf16.h>
#include <math.h>

// ---- problem constants (match reference) ----
#define BATCH 128
#define SEQ   80
#define VOCAB 10000
#define EMB   100
#define EMBP  128      // EMB padded to multiple of 32 for WMMA K
#define UNITS 1024
#define NBLK  64       // persistent grid: 8 m-tiles x 8 n-blocks

typedef __attribute__((ext_vector_type(16))) _Float16 v16h;
typedef __attribute__((ext_vector_type(8)))  _Float16 v8h;
typedef __attribute__((ext_vector_type(8)))  float    v8f;

// ------------------------------------------------------------------
// WMMA fragment loads, layouts per CDNA5 ISA 7.12.2 (wave32).
// A (16x32 f16, row-major source, stride lda):
//   lane l: row = m0 + (l&15); elems 0..7  -> K = kb + 8*(l>>4) + e
//                              elems 8..15 -> K = kb + 8*(l>>4) + 16 + (e-8)
// B (32x16 f16) loaded from PRE-TRANSPOSED weights Bt[N][K]:
//   lane l: col = n0 + (l&15); elems 0..7  -> K = kb + 8*(l>>4) + e
//                              elems 8..15 -> K = kb + 8*(l>>4) + 16 + (e-8)
// C/D (16x16 f32): VGPR i <-> row m0 + 8*(l>>4) + i ; lane&15 <-> col
// ------------------------------------------------------------------
__device__ __forceinline__ v16h load_fragA(const _Float16* __restrict__ A, int lda,
                                           int m0, int kb, int lane) {
  const int row = m0 + (lane & 15);
  const _Float16* p = A + (size_t)row * lda + kb + ((lane >> 4) << 3);
  v8h lo = *(const v8h*)(p);
  v8h hi = *(const v8h*)(p + 16);
  return __builtin_shufflevector(lo, hi, 0,1,2,3,4,5,6,7,8,9,10,11,12,13,14,15);
}

__device__ __forceinline__ v16h load_fragB(const _Float16* __restrict__ Bt, int ldk,
                                           int n0, int kb, int lane) {
  const int col = n0 + (lane & 15);
  const _Float16* p = Bt + (size_t)col * ldk + kb + ((lane >> 4) << 3);
  v8h lo = *(const v8h*)(p);
  v8h hi = *(const v8h*)(p + 16);
  return __builtin_shufflevector(lo, hi, 0,1,2,3,4,5,6,7,8,9,10,11,12,13,14,15);
}

#define WMMA_F16(a, b, c) \
  __builtin_amdgcn_wmma_f32_16x16x32_f16(false, (a), false, (b), (short)0, (c), false, false)

// ------------------------------------------------------------------
// one-time prep kernels
// ------------------------------------------------------------------

// W[K,N] f32 -> Wt[N,Kpad] f16 (zero-padded K)
__global__ void __launch_bounds__(256)
k_transpose_w(const float* __restrict__ W, _Float16* __restrict__ Wt,
              int K, int N, int Kpad) {
  int idx = blockIdx.x * 256 + threadIdx.x;
  if (idx >= N * Kpad) return;
  int n = idx / Kpad, k = idx - n * Kpad;
  float v = (k < K) ? W[(size_t)k * N + n] : 0.0f;
  Wt[(size_t)n * Kpad + k] = (_Float16)v;
}

// Embedding gather -> time-major padded f16 X [SEQ*BATCH, EMBP]
__global__ void __launch_bounds__(EMBP)
k_embed(const int* __restrict__ inputs, const float* __restrict__ emb,
        _Float16* __restrict__ X) {
  int row = blockIdx.x;                 // row = t*BATCH + b
  int t = row / BATCH, b = row - t * BATCH;
  int e = threadIdx.x;
  int tok = inputs[b * SEQ + t];
  float v = (e < EMB) ? emb[(size_t)tok * EMB + e] : 0.0f;
  X[(size_t)row * EMBP + e] = (_Float16)v;
}

__global__ void __launch_bounds__(256)
k_zero_h(_Float16* __restrict__ p, int n) {
  int i = blockIdx.x * 256 + threadIdx.x;
  if (i < n) p[i] = (_Float16)0.0f;
}

__global__ void __launch_bounds__(64)
k_zero_u32(unsigned* __restrict__ p) {
  if (threadIdx.x == 0 && blockIdx.x == 0) *p = 0u;
}

// XK = X @ Wk0 + b0   (M = SEQ*BATCH, K = EMBP, N = UNITS), f32 out
__global__ void __launch_bounds__(256)
k_gemm_xk(const _Float16* __restrict__ X, const _Float16* __restrict__ Wk0t,
          const float* __restrict__ b0, float* __restrict__ XK) {
  const int lane = threadIdx.x & 31, w = threadIdx.x >> 5;
  const int m0 = blockIdx.y * 16;
  const int n0 = blockIdx.x * 128 + w * 16;
  const int n = lane & 15, mb = (lane >> 4) * 8;
  const float bias = b0[n0 + n];
  v8f c0, c1;
  #pragma unroll
  for (int i = 0; i < 8; ++i) { c0[i] = bias; c1[i] = 0.0f; }
  #pragma unroll
  for (int kb = 0; kb < EMBP; kb += 64) {          // dual split-K chains
    v16h a0 = load_fragA(X, EMBP, m0, kb, lane);
    v16h bb0 = load_fragB(Wk0t, EMBP, n0, kb, lane);
    v16h a1 = load_fragA(X, EMBP, m0, kb + 32, lane);
    v16h bb1 = load_fragB(Wk0t, EMBP, n0, kb + 32, lane);
    c0 = WMMA_F16(a0, bb0, c0);
    c1 = WMMA_F16(a1, bb1, c1);
  }
  #pragma unroll
  for (int i = 0; i < 8; ++i)
    XK[(size_t)(m0 + mb + i) * UNITS + n0 + n] = c0[i] + c1[i];
}

// ------------------------------------------------------------------
// device-wide barrier for the persistent kernel (monotonic counter,
// zeroed by k_zero_u32 before each launch -> deterministic replays)
// ------------------------------------------------------------------
__device__ __forceinline__ void gsync(unsigned* __restrict__ cnt, unsigned target) {
  __threadfence();                       // release this block's stores
  __syncthreads();
  if (threadIdx.x == 0) {
    atomicAdd(cnt, 1u);
    while (*(volatile unsigned*)cnt < target) {
      __builtin_amdgcn_s_sleep(1);
    }
  }
  __syncthreads();
  __threadfence();                       // acquire side
}

// ------------------------------------------------------------------
// persistent RNN scan: 64 blocks run all SEQ timesteps.
// Block b: m0 = (b>>3)*16 ; per wave n0 = (b&7)*128 + wave*16.
// One grid barrier per timestep (between layer0 and layer1); the
// layer1 -> next layer0 ordering is covered transitively because a
// block signals barrier t only after finishing its layer1 of t-1.
// ------------------------------------------------------------------
__global__ void __launch_bounds__(256)
k_rnn_scan(const float* __restrict__ XK,
           const _Float16* __restrict__ Wr0t, const _Float16* __restrict__ Wk1t,
           const _Float16* __restrict__ Wr1t, const float* __restrict__ b1c,
           _Float16* __restrict__ h0a, _Float16* __restrict__ h0b,
           _Float16* __restrict__ h1a, _Float16* __restrict__ h1b,
           const _Float16* __restrict__ Wd1t, const float* __restrict__ bd1,
           float* __restrict__ z, unsigned* __restrict__ bar) {
  const int lane = threadIdx.x & 31, w = threadIdx.x >> 5;
  const int m0 = (blockIdx.x >> 3) * 16;
  const int n0 = (blockIdx.x & 7) * 128 + w * 16;
  const int n = lane & 15, mb = (lane >> 4) * 8;

  _Float16* h0buf[2] = {h0a, h0b};
  _Float16* h1buf[2] = {h1a, h1b};
  unsigned phase = 0;

  for (int t = 0; t < SEQ; ++t) {
    const int cur = t & 1;
    // ---------- layer0: h0' = tanh(XK_t + h0 @ Wr0) ----------
    {
      const float* XKt = XK + (size_t)t * BATCH * UNITS;
      const _Float16* A = h0buf[cur];
      v8f c0, c1;
      #pragma unroll
      for (int i = 0; i < 8; ++i) {
        c0[i] = XKt[(size_t)(m0 + mb + i) * UNITS + n0 + n];
        c1[i] = 0.0f;
      }
      #pragma unroll 4
      for (int kb = 0; kb < UNITS; kb += 64) {     // dual split-K chains
        __builtin_prefetch(Wr0t + (size_t)(n0 + n) * UNITS + kb + 128, 0, 1);
        v16h a0 = load_fragA(A, UNITS, m0, kb, lane);
        v16h bb0 = load_fragB(Wr0t, UNITS, n0, kb, lane);
        v16h a1 = load_fragA(A, UNITS, m0, kb + 32, lane);
        v16h bb1 = load_fragB(Wr0t, UNITS, n0, kb + 32, lane);
        c0 = WMMA_F16(a0, bb0, c0);
        c1 = WMMA_F16(a1, bb1, c1);
      }
      _Float16* H = h0buf[1 - cur];
      #pragma unroll
      for (int i = 0; i < 8; ++i)
        H[(size_t)(m0 + mb + i) * UNITS + n0 + n] = (_Float16)tanhf(c0[i] + c1[i]);
    }
    gsync(bar, (++phase) * NBLK);                  // h0' visible device-wide
    // ---------- layer1: h1' = tanh(h0' @ Wk1 + h1 @ Wr1 + b1c) ----------
    {
      const _Float16* A0 = h0buf[1 - cur];
      const _Float16* A1 = h1buf[cur];
      v8f c0, c1;
      const float bias = b1c[n0 + n];
      #pragma unroll
      for (int i = 0; i < 8; ++i) { c0[i] = bias; c1[i] = 0.0f; }
      #pragma unroll 4
      for (int kb = 0; kb < UNITS; kb += 32) {     // two GEMMs interleaved
        __builtin_prefetch(Wk1t + (size_t)(n0 + n) * UNITS + kb + 64, 0, 1);
        __builtin_prefetch(Wr1t + (size_t)(n0 + n) * UNITS + kb + 64, 0, 1);
        v16h a0 = load_fragA(A0, UNITS, m0, kb, lane);
        v16h bb0 = load_fragB(Wk1t, UNITS, n0, kb, lane);
        v16h a1 = load_fragA(A1, UNITS, m0, kb, lane);
        v16h bb1 = load_fragB(Wr1t, UNITS, n0, kb, lane);
        c0 = WMMA_F16(a0, bb0, c0);
        c1 = WMMA_F16(a1, bb1, c1);
      }
      _Float16* H = h1buf[1 - cur];
      #pragma unroll
      for (int i = 0; i < 8; ++i)
        H[(size_t)(m0 + mb + i) * UNITS + n0 + n] = (_Float16)tanhf(c0[i] + c1[i]);
    }
  }
  gsync(bar, (++phase) * NBLK);                    // final h1 visible
  // ---------- head GEMM: z = relu(h1 @ Wd1 + bd1) ----------
  {
    const _Float16* A = h1buf[SEQ & 1];
    v8f c0, c1;
    const float bias = bd1[n0 + n];
    #pragma unroll
    for (int i = 0; i < 8; ++i) { c0[i] = bias; c1[i] = 0.0f; }
    #pragma unroll 4
    for (int kb = 0; kb < UNITS; kb += 64) {
      v16h a0 = load_fragA(A, UNITS, m0, kb, lane);
      v16h bb0 = load_fragB(Wd1t, UNITS, n0, kb, lane);
      v16h a1 = load_fragA(A, UNITS, m0, kb + 32, lane);
      v16h bb1 = load_fragB(Wd1t, UNITS, n0, kb + 32, lane);
      c0 = WMMA_F16(a0, bb0, c0);
      c1 = WMMA_F16(a1, bb1, c1);
    }
    #pragma unroll
    for (int i = 0; i < 8; ++i)
      z[(size_t)(m0 + mb + i) * UNITS + n0 + n] = fmaxf(c0[i] + c1[i], 0.0f);
  }
}

// out[m] = sigmoid(z[m,:] . Wd2 + bd2)   -- one block per batch row
__global__ void __launch_bounds__(256)
k_final(const float* __restrict__ z, const float* __restrict__ Wd2,
        const float* __restrict__ bd2, float* __restrict__ out) {
  __shared__ float red[256];
  const int m = blockIdx.x;
  float s = 0.0f;
  for (int k = threadIdx.x; k < UNITS; k += 256)
    s += z[(size_t)m * UNITS + k] * Wd2[k];
  red[threadIdx.x] = s;
  __syncthreads();
  for (int off = 128; off > 0; off >>= 1) {
    if (threadIdx.x < off) red[threadIdx.x] += red[threadIdx.x + off];
    __syncthreads();
  }
  if (threadIdx.x == 0) {
    float y = red[0] + bd2[0];
    out[m] = 1.0f / (1.0f + __expf(-y));
  }
}

// ------------------------------------------------------------------
extern "C" void kernel_launch(void* const* d_in, const int* in_sizes, int n_in,
                              void* d_out, int out_size, void* d_ws, size_t ws_size,
                              hipStream_t stream) {
  const int*   inputs = (const int*)  d_in[0];
  const float* emb    = (const float*)d_in[1];
  const float* Wk0    = (const float*)d_in[2];
  const float* Wr0    = (const float*)d_in[3];
  const float* b0     = (const float*)d_in[4];
  const float* Wk1    = (const float*)d_in[5];
  const float* Wr1    = (const float*)d_in[6];
  const float* b1c    = (const float*)d_in[7];
  const float* Wd1    = (const float*)d_in[8];
  const float* bd1    = (const float*)d_in[9];
  const float* Wd2    = (const float*)d_in[10];
  const float* bd2    = (const float*)d_in[11];
  float* out = (float*)d_out;

  char* ws = (char*)d_ws;
  auto alloc = [&](size_t bytes) {
    char* p = ws;
    ws += (bytes + 255) & ~(size_t)255;
    return p;
  };

  _Float16* Wk0t = (_Float16*)alloc((size_t)UNITS * EMBP  * 2);
  _Float16* Wr0t = (_Float16*)alloc((size_t)UNITS * UNITS * 2);
  _Float16* Wk1t = (_Float16*)alloc((size_t)UNITS * UNITS * 2);
  _Float16* Wr1t = (_Float16*)alloc((size_t)UNITS * UNITS * 2);
  _Float16* Wd1t = (_Float16*)alloc((size_t)UNITS * UNITS * 2);
  _Float16* X    = (_Float16*)alloc((size_t)SEQ * BATCH * EMBP * 2);
  float*    XK   = (float*)   alloc((size_t)SEQ * BATCH * UNITS * 4);
  _Float16* h0a  = (_Float16*)alloc((size_t)BATCH * UNITS * 2);
  _Float16* h0b  = (_Float16*)alloc((size_t)BATCH * UNITS * 2);
  _Float16* h1a  = (_Float16*)alloc((size_t)BATCH * UNITS * 2);
  _Float16* h1b  = (_Float16*)alloc((size_t)BATCH * UNITS * 2);
  float*    zbuf = (float*)   alloc((size_t)BATCH * UNITS * 4);
  unsigned* bar  = (unsigned*)alloc(256);

  // --- weight convert + transpose (f32 [K,N] -> f16 [N,Kpad]) ---
  {
    int nel = UNITS * EMBP;
    k_transpose_w<<<(nel + 255) / 256, 256, 0, stream>>>(Wk0, Wk0t, EMB, UNITS, EMBP);
    nel = UNITS * UNITS;
    int blocks = (nel + 255) / 256;
    k_transpose_w<<<blocks, 256, 0, stream>>>(Wr0, Wr0t, UNITS, UNITS, UNITS);
    k_transpose_w<<<blocks, 256, 0, stream>>>(Wk1, Wk1t, UNITS, UNITS, UNITS);
    k_transpose_w<<<blocks, 256, 0, stream>>>(Wr1, Wr1t, UNITS, UNITS, UNITS);
    k_transpose_w<<<blocks, 256, 0, stream>>>(Wd1, Wd1t, UNITS, UNITS, UNITS);
  }

  // --- embedding gather + hoisted input projection ---
  k_embed<<<SEQ * BATCH, EMBP, 0, stream>>>(inputs, emb, X);
  k_gemm_xk<<<dim3(UNITS / 128, (SEQ * BATCH) / 16), 256, 0, stream>>>(X, Wk0t, b0, XK);

  // --- zero-init hidden states + barrier counter ---
  {
    int nh = BATCH * UNITS;
    k_zero_h<<<(nh + 255) / 256, 256, 0, stream>>>(h0a, nh);
    k_zero_h<<<(nh + 255) / 256, 256, 0, stream>>>(h1a, nh);
    k_zero_u32<<<1, 64, 0, stream>>>(bar);
  }

  // --- persistent scan over all timesteps + head GEMM ---
  k_rnn_scan<<<NBLK, 256, 0, stream>>>(XK, Wr0t, Wk1t, Wr1t, b1c,
                                       h0a, h0b, h1a, h1b,
                                       Wd1t, bd1, zbuf, bar);

  // --- final dot + sigmoid ---
  k_final<<<BATCH, 256, 0, stream>>>(zbuf, Wd2, bd2, out);
}
